// PnPNystraAttention_21509196218786
// MI455X (gfx1250) — matrix operand order repaired and checked by reading, not compiled, not run on hardware
//
#include <hip/hip_runtime.h>
#include <hip/hip_bf16.h>
#include <math.h>

typedef __attribute__((ext_vector_type(16))) _Float16 v16h;
typedef __attribute__((ext_vector_type(8)))  float    v8f;

#define DEV static __device__ __attribute__((always_inline)) inline

DEV float activf(float x) {
    // exp(min(x,5)) + relu(x-5)
    return __expf(fminf(x, 5.0f)) + fmaxf(x - 5.0f, 0.0f);
}

// ---------------- WMMA fragment loaders (wave32, 16x16x32 f16) ----------------
// A fragment: rows m0..m0+15, K = k0..k0+31 of row-major f16 buffer.
// Lane L: m = m0 + (L&15), h = L>>4. v16h element e maps to
// K = k0 + h*16 + e  => one contiguous 32B run.
DEV v16h load_a(const _Float16* buf, int pitch, int m0, int k0, int lane) {
    int m = m0 + (lane & 15);
    int h = lane >> 4;
    const _Float16* p = buf + m * pitch + k0 + h * 16;
    union { uint4 u[2]; v16h v; } f;
    f.u[0] = *(const uint4*)(p);
    f.u[1] = *(const uint4*)(p + 8);
    return f.v;
}

// Same A layout but source is row-major f32 in global memory (convert on the fly).
DEV v16h load_a_g32(const float* buf, int pitch, int m0, int k0, int lane) {
    int m = m0 + (lane & 15);
    int h = lane >> 4;
    const float* p = buf + m * pitch + k0 + h * 16;
    v16h a;
#pragma unroll
    for (int e = 0; e < 8; ++e) a[e] = (_Float16)p[e];
#pragma unroll
    for (int e = 0; e < 8; ++e) a[8 + e] = (_Float16)p[8 + e];
    return a;
}

// B fragment for D = A*B^T: "B" operand built from rows n0..n0+15 of a row-major
// buffer (K runs along the row). Lane L: n = n0+(L&15), h = L>>4,
// element e <-> K = k0 + h*16 + e  => one contiguous 32B run.
DEV v16h load_bt(const _Float16* buf, int pitch, int n0, int k0, int lane) {
    int n = n0 + (lane & 15);
    int h = lane >> 4;
    const _Float16* p = buf + n * pitch + k0 + h * 16;
    union { uint4 u[2]; v16h v; } f;
    f.u[0] = *(const uint4*)(p);
    f.u[1] = *(const uint4*)(p + 8);
    return f.v;
}

// Same but source rows are f32 in global memory.
DEV v16h load_bt_g32(const float* buf, int pitch, int n0, int k0, int lane) {
    int n = n0 + (lane & 15);
    int h = lane >> 4;
    const float* p = buf + n * pitch + k0 + h * 16;
    v16h b;
#pragma unroll
    for (int e = 0; e < 16; ++e) b[e] = (_Float16)p[e];
    return b;
}

#define WMMA(a, b, c) \
    __builtin_amdgcn_wmma_f32_16x16x32_f16(false, (a), false, (b), (short)0, (c), false, false)

// ---------------- tiled GEMM micro-kernels (one 16x16 output tile) ----------------
DEV v8f mm_ABt(const _Float16* A, int pa, const _Float16* B, int pb,
               int m0, int n0, int K, int lane) {
    v8f acc = {};
    for (int k0 = 0; k0 < K; k0 += 32)
        acc = WMMA(load_a(A, pa, m0, k0, lane), load_bt(B, pb, n0, k0, lane), acc);
    return acc;
}
DEV v8f mm_ABt_gA(const float* A, int pa, const _Float16* B, int pb,
                  int m0, int n0, int K, int lane) {
    v8f acc = {};
    for (int k0 = 0; k0 < K; k0 += 32)
        acc = WMMA(load_a_g32(A, pa, m0, k0, lane), load_bt(B, pb, n0, k0, lane), acc);
    return acc;
}
DEV v8f mm_ABt_gB(const _Float16* A, int pa, const float* B, int pb,
                  int m0, int n0, int K, int lane) {
    v8f acc = {};
    for (int k0 = 0; k0 < K; k0 += 32)
        acc = WMMA(load_a(A, pa, m0, k0, lane), load_bt_g32(B, pb, n0, k0, lane), acc);
    return acc;
}

// tile store epilogues (identity / aI - X / 0.25x / activ)
enum { ST_ID = 0, ST_7I = 1, ST_15I = 2, ST_13I = 3, ST_QZ = 4, ST_ACT = 5 };

DEV float epilogue(float v, int m, int n, int mode) {
    switch (mode) {
        default:     return v;
        case ST_7I:  return (m == n ? 7.0f  : 0.0f) - v;
        case ST_15I: return (m == n ? 15.0f : 0.0f) - v;
        case ST_13I: return (m == n ? 13.0f : 0.0f) - v;
        case ST_QZ:  return 0.25f * v;
        case ST_ACT: return activf(v);
    }
}

// Store tile in natural orientation: dst[m*pitch + n]. 8 ds_store_b16 per lane.
DEV void store_tile(_Float16* dst, int pitch, int m0, int n0, v8f acc, int lane, int mode) {
    int n = n0 + (lane & 15);
    int h = lane >> 4;
#pragma unroll
    for (int r = 0; r < 8; ++r) {
        int m = m0 + r + 8 * h;
        dst[m * pitch + n] = (_Float16)epilogue(acc[r], m, n, mode);
    }
}

// Store tile TRANSPOSED: dst[n*pitch + m]. Each lane's 8 values are contiguous
// along m -> single 16-byte ds_store_b128 per lane.
DEV void store_tile_T(_Float16* dst, int pitch, int m0, int n0, v8f acc, int lane, int mode) {
    int n = n0 + (lane & 15);
    int h = lane >> 4;
    int mb = m0 + 8 * h;
    union { _Float16 hh[8]; uint4 u; } pk;
#pragma unroll
    for (int r = 0; r < 8; ++r)
        pk.hh[r] = (_Float16)epilogue(acc[r], mb + r, n, mode);
    *(uint4*)(dst + n * pitch + mb) = pk.u;
}

// ---------------- fused Nystrom attention: one workgroup per (b,h) ----------------
__global__ __launch_bounds__(256, 1)
void nystra_fused_kernel(const float* __restrict__ Q, const float* __restrict__ K,
                         const float* __restrict__ V, float* __restrict__ O) {
    // LDS budget ~155 KB (< 160 KB)
    __shared__ __align__(16) _Float16 sQm[64 * 64];    // pooled q landmarks
    __shared__ __align__(16) _Float16 sKm[64 * 64];    // pooled k landmarks
    __shared__ __align__(16) _Float16 sX [64 * 64];    // activ(qm km^T)
    __shared__ __align__(16) _Float16 sB0[64 * 64];    // pinv rotating buffers
    __shared__ __align__(16) _Float16 sB1[64 * 64];
    __shared__ __align__(16) _Float16 sB2[64 * 64];
    __shared__ __align__(16) _Float16 sB3[64 * 64];
    __shared__ __align__(16) _Float16 sTt[64 * 64];    // transposed T operand
    __shared__ __align__(16) _Float16 sS2[64 * 256];   // activ(qm k^T); reused as qk [256x64]
    __shared__ __align__(16) _Float16 sKVt[80 * 64];   // kv^T (row 64 = ones column, 65..79 zero)
    __shared__ __align__(16) _Float16 sVt[64 * 256];   // v^T in f16
    __shared__ __align__(16) _Float16 sP [8 * 16 * 64];// per-wave p-block scratch
    __shared__ float sRed[128];
    __shared__ float sDen[8][16];
    __shared__ float sScale;

    const int bh   = blockIdx.x;
    const long base = (long)bh * 256 * 64;
    const float* gq = Q + base;
    const float* gk = K + base;
    const float* gv = V + base;
    float*       go = O + base;

    const int tid  = threadIdx.x;
    const int w    = tid >> 5;
    const int lane = tid & 31;

    // ---- stage 0: v -> LDS transposed f16; landmark 2x2 average pooling of q,k ----
    for (int i = tid; i < 256 * 64; i += 256) {
        int n = i >> 6, c = i & 63;                 // coalesced global read
        sVt[c * 256 + n] = (_Float16)gv[i];
    }
    for (int i = tid; i < 64 * 64; i += 256) {
        int l = i >> 6, dd = i & 63;
        int lh = l >> 3, lw = l & 7;
        int r0 = (2 * lh) * 16 + 2 * lw;            // n = (2lh+ii)*16 + 2lw+jj
        float sq = 0.f, sk = 0.f;
#pragma unroll
        for (int ii = 0; ii < 2; ++ii)
#pragma unroll
            for (int jj = 0; jj < 2; ++jj) {
                int n = r0 + ii * 16 + jj;
                sq += gq[n * 64 + dd];
                sk += gk[n * 64 + dd];
            }
        sQm[i] = (_Float16)(sq * 0.25f);
        sKm[i] = (_Float16)(sk * 0.25f);
    }
    __syncthreads();

    // ---- stage 1: X = activ(qm km^T)  [64x64], 16 tiles / 8 waves ----
    for (int t = w; t < 16; t += 8) {
        int m0 = (t >> 2) * 16, n0 = (t & 3) * 16;
        v8f acc = mm_ABt(sQm, 64, sKm, 64, m0, n0, 64, lane);
        store_tile(sX, 64, m0, n0, acc, lane, ST_ACT);
    }
    __syncthreads();

    // ---- stage 2: pinv init  z = X^T / (max row-abs-sum * max col-abs-sum) ----
    if (tid < 128) {
        int j = tid & 63;
        float s = 0.f;
        if (tid < 64) { for (int c = 0; c < 64; ++c)  s += fabsf((float)sX[j * 64 + c]); }
        else          { for (int rr = 0; rr < 64; ++rr) s += fabsf((float)sX[rr * 64 + j]); }
        sRed[tid] = s;
    }
    __syncthreads();
    if (tid == 0) {
        float mr = 0.f, mc = 0.f;
        for (int j = 0; j < 64; ++j) { mr = fmaxf(mr, sRed[j]); mc = fmaxf(mc, sRed[64 + j]); }
        sScale = 1.0f / (mr * mc + 1e-15f);
    }
    __syncthreads();
    // Zn = X^T * scale (natural), Zt = Z^T = X * scale
    for (int i = tid; i < 64 * 64; i += 256) {
        int m = i >> 6, n = i & 63;
        float xv = (float)sX[i] * sScale;
        sB1[i] = (_Float16)xv;                      // Zt
        sB0[n * 64 + m] = (_Float16)xv;             // Zn
    }
    __syncthreads();

    // ---- stage 3: S2 = activ(qm k^T)  [64x256], k rows streamed from global ----
    for (int t = w; t < 64; t += 8) {
        int m0 = (t & 3) * 16, n0 = (t >> 2) * 16;
        v8f acc = mm_ABt_gB(sQm, 64, gk, 64, m0, n0, 64, lane);
        store_tile(sS2, 256, m0, n0, acc, lane, ST_ACT);
    }
    __syncthreads();

    // ---- stage 4: kv^T = (S2 * [v | 1])^T  -> sKVt [80x64] ----
    for (int t = w; t < 16; t += 8) {
        int m0 = (t & 3) * 16, n0 = (t >> 2) * 16;
        v8f acc = mm_ABt(sS2, 256, sVt, 256, m0, n0, 256, lane);
        store_tile_T(sKVt, 64, m0, n0, acc, lane, ST_ID);
    }
    if (tid < 64) {                       // ones-column row (kv[:,64]) + zero padding rows
        float s = 0.f;
        for (int c = 0; c < 256; ++c) s += (float)sS2[tid * 256 + c];
        sKVt[64 * 64 + tid] = (_Float16)s;
        for (int rr = 65; rr < 80; ++rr) sKVt[rr * 64 + tid] = (_Float16)0.0f;
    }
    __syncthreads();

    // ---- stage 5: qk = activ(q km^T)  [256x64], overwrites sS2 ----
    _Float16* qkb = sS2;
    for (int t = w; t < 64; t += 8) {
        int m0 = (t >> 2) * 16, n0 = (t & 3) * 16;
        v8f acc = mm_ABt_gA(gq, 64, sKm, 64, m0, n0, 64, lane);
        store_tile(qkb, 64, m0, n0, acc, lane, ST_ACT);
    }
    __syncthreads();

    // ---- stage 6: Newton-Schulz pinv, 6 iterations, all matmuls in A*B^T form ----
    // Rotation: {Zn, Zt, Fa, Fb} <- {Fa, Fb, Zn, Zt}; sTt holds the T operand (transposed).
    _Float16 *Zn = sB0, *Zt = sB1, *Fa = sB2, *Fb = sB3;
    for (int it = 0; it < 6; ++it) {
        for (int t = w; t < 16; t += 8) {           // XZ = X*Z (= X*(Zt)^T); Tt = (7I-XZ)^T
            int m0 = (t >> 2) * 16, n0 = (t & 3) * 16;
            v8f acc = mm_ABt(sX, 64, Zt, 64, m0, n0, 64, lane);
            store_tile  (Fa,  64, m0, n0, acc, lane, ST_ID);
            store_tile_T(sTt, 64, m0, n0, acc, lane, ST_7I);
        }
        __syncthreads();
        for (int t = w; t < 16; t += 8) {           // Ut = (15I - XZ*T)^T
            int m0 = (t >> 2) * 16, n0 = (t & 3) * 16;
            v8f acc = mm_ABt(Fa, 64, sTt, 64, m0, n0, 64, lane);
            store_tile_T(Fb, 64, m0, n0, acc, lane, ST_15I);
        }
        __syncthreads();
        for (int t = w; t < 16; t += 8) {           // Tt = (13I - XZ*U)^T
            int m0 = (t >> 2) * 16, n0 = (t & 3) * 16;
            v8f acc = mm_ABt(Fa, 64, Fb, 64, m0, n0, 64, lane);
            store_tile_T(sTt, 64, m0, n0, acc, lane, ST_13I);
        }
        __syncthreads();
        for (int t = w; t < 16; t += 8) {           // Zout = 0.25 * Z*T (both orientations)
            int m0 = (t >> 2) * 16, n0 = (t & 3) * 16;
            v8f acc = mm_ABt(Zn, 64, sTt, 64, m0, n0, 64, lane);
            store_tile  (Fa, 64, m0, n0, acc, lane, ST_QZ);
            store_tile_T(Fb, 64, m0, n0, acc, lane, ST_QZ);
        }
        __syncthreads();
        _Float16* t0 = Zn; _Float16* t1 = Zt;
        Zn = Fa; Zt = Fb; Fa = t0; Fb = t1;
    }
    const _Float16* Zft = Zt;                       // pinv^T

    // ---- stage 7: out = rownorm( (qk * pinv) * kv ), 16-row blocks, 2 per wave ----
    _Float16* pblk = sP + w * 16 * 64;
    for (int rb = w; rb < 16; rb += 8) {
        // p = qk_block(16x64) * Z = qk*(Zft)^T -> per-wave LDS scratch (same-wave LDS in-order)
        for (int cb = 0; cb < 4; ++cb) {
            v8f acc = mm_ABt(qkb, 64, Zft, 64, rb * 16, cb * 16, 64, lane);
            store_tile(pblk, 64, 0, cb * 16, acc, lane, ST_ID);
        }
        // normalizer column: p * kv[:,64..79] = p * (kvT rows 64..79)^T (row 64 real, rest zero)
        {
            v8f acc = mm_ABt(pblk, 64, sKVt, 64, 0, 64, 64, lane);
            if ((lane & 15) == 0) {
                int h = lane >> 4;
#pragma unroll
                for (int r = 0; r < 8; ++r) sDen[w][r + 8 * h] = acc[r] + 1e-12f;
            }
        }
        // value columns: divide by normalizer, write f32 to global
        for (int cb = 0; cb < 4; ++cb) {
            v8f acc = mm_ABt(pblk, 64, sKVt, 64, 0, cb * 16, 64, lane);
            int n = cb * 16 + (lane & 15);
            int h = lane >> 4;
#pragma unroll
            for (int r = 0; r < 8; ++r) {
                int m = rb * 16 + r + 8 * h;
                go[m * 64 + n] = acc[r] / sDen[w][r + 8 * h];
            }
        }
    }
}

extern "C" void kernel_launch(void* const* d_in, const int* in_sizes, int n_in,
                              void* d_out, int out_size, void* d_ws, size_t ws_size,
                              hipStream_t stream) {
    const float* q = (const float*)d_in[0];
    const float* k = (const float*)d_in[1];
    const float* v = (const float*)d_in[2];
    float* out = (float*)d_out;
    (void)in_sizes; (void)n_in; (void)out_size; (void)d_ws; (void)ws_size;

    dim3 grid(256 * 16);   // one workgroup per (b, h)
    dim3 block(256);       // 8 wave32 waves
    nystra_fused_kernel<<<grid, block, 0, stream>>>(q, k, v, out);
}
